// MultiZoomSelfAttention_12008728559999
// MI455X (gfx1250) — matrix operand order repaired and checked by reading, not compiled, not run on hardware
//
#include <hip/hip_runtime.h>
#include <hip/hip_bf16.h>
#include <math.h>

// ---- problem constants (from reference) ----
#define NHEAD   16
#define HDIM    16
#define CDIM    256      // NH*HD
#define EDIM    64
#define TWO_C   512
#define NTOK    64       // n
#define NGRP    2048     // b*v*t*s
#define NROWS   (NGRP * NTOK)   // 131072
#define NEGVAL  (-1000000000.0f)

typedef __attribute__((ext_vector_type(16))) _Float16 v16h;
typedef __attribute__((ext_vector_type(8)))  float    v8f;
typedef __attribute__((ext_vector_type(4)))  unsigned int v4u;
typedef __attribute__((ext_vector_type(8)))  int      v8i;
typedef __attribute__((ext_vector_type(4)))  int      v4i;

#if __has_builtin(__builtin_amdgcn_tensor_load_to_lds)
#define HAVE_TDM 1
#else
#define HAVE_TDM 0
#endif

#define WMMA_F32_F16(A, B, C) \
  __builtin_amdgcn_wmma_f32_16x16x32_f16(false, (A), false, (B), (short)0, (C), false, false)

// A-fragment (16x32 f16): lane holds row M=lane%16; element e maps to
// K = e + 8*(lane/16) + 8*(e>>3)   (within the 32-wide K chunk)
__device__ __forceinline__ int a_k(int e, int hg) { return e + 8 * hg + 8 * (e >> 3); }
// B-fragment (32x16 f16): lane holds col N=lane%16; element e maps to
// K = e + 8*(lane/16) + 8*(e>>3)
__device__ __forceinline__ int b_k(int e, int hg) { return e + 8 * hg + 8 * (e >> 3); }

// gate = 1 + 0.5*tanh(y): single trans op if the HW tanh builtin exists,
// else branch-free exp form with raw v_rcp_f32 (no IEEE-div fixup chain).
__device__ __forceinline__ float gate_of(float y) {
#if __has_builtin(__builtin_amdgcn_tanhf)
  return 1.0f + 0.5f * __builtin_amdgcn_tanhf(y);
#else
  return 1.5f - __builtin_amdgcn_rcpf(__expf(2.0f * y) + 1.0f);
#endif
}

// ---------------------------------------------------------------------------
// Kernel 0: mask reduction.  mask_out[g] = AND over n; m = mask_out ? 0 : mask
// ---------------------------------------------------------------------------
__global__ __launch_bounds__(64) void mask_kernel(const unsigned char* __restrict__ mask,
                                                  unsigned char* __restrict__ maskm,
                                                  float* __restrict__ mask_out_f) {
  int g = blockIdx.x;
  int t = threadIdx.x;   // 0..63
  __shared__ int allm;
  if (t == 0) allm = 1;
  __syncthreads();
  unsigned char mv = mask[g * NTOK + t];
  if (!mv) atomicAnd(&allm, 0);
  __syncthreads();
  int mo = allm;
  maskm[g * NTOK + t] = mo ? (unsigned char)0 : mv;
  mask_out_f[g * NTOK + t] = mo ? 1.0f : 0.0f;
}

// ---------------------------------------------------------------------------
// Kernel 1: q = (emb_q @ W_emb + b_emb) * (1 + 0.5*tanh(emb_q @ W_gate + b_gate))
// One wave per 16 rows.  Output Qh as f16 in [G][NH][n][HD] layout.
// ---------------------------------------------------------------------------
__global__ __launch_bounds__(256) void qproj_kernel(const float* __restrict__ emb_q,
                                                    const float* __restrict__ W_emb,
                                                    const float* __restrict__ b_emb,
                                                    const float* __restrict__ W_gate,
                                                    const float* __restrict__ b_gate,
                                                    _Float16* __restrict__ Qh) {
  const int lane = threadIdx.x & 31;
  const int wid  = threadIdx.x >> 5;
  const int rb   = blockIdx.x * 8 + wid;    // 0..8191 row-blocks
  const int row0 = rb * 16;
  const int m    = lane & 15;
  const int hg   = lane >> 4;
  const int rowm = row0 + m;

  // A fragments: 16 rows of emb_q (K = 64 -> 2 chunks of 32)
  v16h a[2];
  for (int kc = 0; kc < 2; ++kc)
    for (int e = 0; e < 16; ++e)
      a[kc][e] = (_Float16)emb_q[(size_t)rowm * EDIM + kc * 32 + a_k(e, hg)];

  // gate GEMM: (16x64) @ (64x16)
  v8f gd = {};
  for (int kc = 0; kc < 2; ++kc) {
    v16h bg;
    for (int e = 0; e < 16; ++e) {
      int k = kc * 32 + b_k(e, hg);
      bg[e] = (_Float16)W_gate[k * NHEAD + m];
    }
    gd = WMMA_F32_F16(a[kc], bg, gd);
  }
  const float bgate = b_gate[m];
  float gval[8];
  for (int r = 0; r < 8; ++r)
    gval[r] = gate_of(gd[r] + bgate);

  const int gidx = row0 >> 6;   // group
  const int tok0 = row0 & 63;   // token base within group

  // q projection: 16 N-tiles; tile nt == head nt (HD == 16)
  for (int nt = 0; nt < 16; ++nt) {
    v8f qd = {};
    for (int kc = 0; kc < 2; ++kc) {
      v16h bw;
      for (int e = 0; e < 16; ++e) {
        int k = kc * 32 + b_k(e, hg);
        bw[e] = (_Float16)W_emb[k * CDIM + nt * 16 + m];
      }
      qd = WMMA_F32_F16(a[kc], bw, qd);
    }
    const float bias = b_emb[nt * 16 + m];
    const int srcl = hg * 16 + nt;   // where g[M][head=nt] lives (ds_bpermute)
    for (int r = 0; r < 8; ++r) {
      float gg = __shfl(gval[r], srcl, 32);
      float q  = (qd[r] + bias) * gg;
      int tok  = tok0 + r + 8 * hg;
      Qh[((size_t)(gidx * NHEAD + nt) * NTOK + tok) * HDIM + m] = (_Float16)q;
    }
  }
}

// ---------------------------------------------------------------------------
// Kernel 2: per (group, head) attention.  One wave per block.
// K/V tile of x (64 tokens x 32 floats, row stride 512 floats) is brought into
// LDS with one Tensor-Data-Mover DMA (TENSOR_LOAD_TO_LDS, TENSORcnt-tracked),
// overlapping the Q-fragment global loads.  scores = QK^T/4 (WMMA, K padded
// 16->32), masked softmax via LDS, out = att @ V (WMMA), f16 to att_out.
// ---------------------------------------------------------------------------
__global__ __launch_bounds__(32) void attn_kernel(const float* __restrict__ x,
                                                  const _Float16* __restrict__ Qh,
                                                  const unsigned char* __restrict__ maskm,
                                                  _Float16* __restrict__ att_out) {
  const int gh = blockIdx.x;          // 0..G*NH-1
  const int g  = gh >> 4;
  const int h  = gh & 15;
  const int lane = threadIdx.x;
  const int m  = lane & 15;
  const int hg = lane >> 4;

  __shared__ float    xs[NTOK][2 * HDIM];   // K|V tile, 8 KiB
  __shared__ float    sc[NTOK][68];
  __shared__ _Float16 at16[NTOK][72];

  const float* gsrc = x + (size_t)(g * NTOK) * TWO_C + h * (2 * HDIM);

#if HAVE_TDM
  {
    // ---- Tensor DMA descriptor (D#), 2D tile: 32 elems x 64 rows, 4B elems
    unsigned long long ga = (unsigned long long)(uintptr_t)gsrc;
    unsigned int lds_off  = (unsigned int)(uintptr_t)&xs[0][0];
    v4u g0;
    g0[0] = 1u;                                  // count=1, user mode
    g0[1] = lds_off;                             // lds_addr (bytes)
    g0[2] = (unsigned int)ga;                    // global_addr[31:0]
    g0[3] = (unsigned int)((ga >> 32) & 0x1FFFFFFu) | (2u << 30); // addr[56:32], type=2
    v8i g1;
    g1[0] = (int)(2u << 16);                     // workgroup_mask=0, data_size=4B
    g1[1] = (int)((2 * HDIM) << 16);             // tensor_dim0[15:0]=32
    g1[2] = (int)(NTOK << 16);                   // tensor_dim0[31:16]=0, tensor_dim1[15:0]=64
    g1[3] = (int)((2 * HDIM) << 16);             // tensor_dim1[31:16]=0, tile_dim0=32
    g1[4] = (int)NTOK;                           // tile_dim1=64, tile_dim2=0
    g1[5] = (int)TWO_C;                          // tensor_dim0_stride[31:0]=512 elems
    g1[6] = 0;                                   // stride0 hi / stride1 lo
    g1[7] = 0;
    v4i z4 = {0, 0, 0, 0};
    v8i z8 = {0, 0, 0, 0, 0, 0, 0, 0};
    __builtin_amdgcn_tensor_load_to_lds(g0, g1, z4, z4, z8, 0);
  }
#else
  for (int t = lane; t < NTOK; t += 32)
    for (int d = 0; d < 2 * HDIM; ++d)
      xs[t][d] = gsrc[(size_t)t * TWO_C + d];
#endif

  // A fragments for Q (overlaps the TDM transfer): 4 row tiles, K pad 16->32
  v16h aq[4];
  for (int rt = 0; rt < 4; ++rt) {
    int tok = rt * 16 + m;
    const _Float16* qp = Qh + ((size_t)(g * NHEAD + h) * NTOK + tok) * HDIM;
    for (int e = 0; e < 16; ++e) {
      int K = a_k(e, hg);
      aq[rt][e] = (K < HDIM) ? qp[K] : (_Float16)0.0f;
    }
  }

#if HAVE_TDM
  __builtin_amdgcn_s_wait_tensorcnt(0);
#endif
  __syncthreads();

  // B fragments for K^T from LDS: 4 column tiles, contraction HD (pad ->32)
  v16h bk[4];
  for (int ct = 0; ct < 4; ++ct) {
    int tok = ct * 16 + m;
    for (int e = 0; e < 16; ++e) {
      int kk = b_k(e, hg);
      bk[ct][e] = (kk < HDIM) ? (_Float16)xs[tok][kk] : (_Float16)0.0f;
    }
  }
  // B fragments for V from LDS: 2 K-chunks of 32 tokens, N = head dim d
  v16h bv[2];
  for (int kc = 0; kc < 2; ++kc)
    for (int e = 0; e < 16; ++e) {
      int tok = kc * 32 + b_k(e, hg);
      bv[kc][e] = (_Float16)xs[tok][HDIM + m];
    }

  // scores -> LDS, with 1/sqrt(HD) scale and key mask
  const unsigned char* mrow = maskm + g * NTOK;
  for (int rt = 0; rt < 4; ++rt)
    for (int ct = 0; ct < 4; ++ct) {
      v8f d = {};
      d = WMMA_F32_F16(aq[rt], bk[ct], d);
      int col = ct * 16 + m;
      bool msk = mrow[col] != 0;
      for (int r = 0; r < 8; ++r) {
        int row = rt * 16 + r + 8 * hg;
        sc[row][col] = msk ? NEGVAL : 0.25f * d[r];
      }
    }
  __syncthreads();

  // row softmax (2 rows per lane), write normalized att as f16
  for (int rr = 0; rr < 2; ++rr) {
    int row = lane + rr * 32;
    float mx = -3.4e38f;
    for (int c = 0; c < NTOK; ++c) mx = fmaxf(mx, sc[row][c]);
    float sum = 0.0f;
    for (int c = 0; c < NTOK; ++c) sum += __expf(sc[row][c] - mx);
    float inv = __builtin_amdgcn_rcpf(sum);
    for (int c = 0; c < NTOK; ++c)
      at16[row][c] = (_Float16)(__expf(sc[row][c] - mx) * inv);
  }
  __syncthreads();

  // out = att @ V
  for (int rt = 0; rt < 4; ++rt) {
    v8f od = {};
    for (int kc = 0; kc < 2; ++kc) {
      v16h aa;
      int row = rt * 16 + m;
      for (int e = 0; e < 16; ++e)
        aa[e] = at16[row][kc * 32 + a_k(e, hg)];
      od = WMMA_F32_F16(aa, bv[kc], od);
    }
    for (int r = 0; r < 8; ++r) {
      int row = rt * 16 + r + 8 * hg;
      att_out[((size_t)(g * NTOK) + row) * CDIM + h * 16 + m] = (_Float16)od[r];
    }
  }
}

// ---------------------------------------------------------------------------
// Kernel 3: out = att_out @ W_out + b_out.  (131072x256)@(256x512), f32 out.
// One wave per 16 rows x 128 cols; A chunks loaded once, reused over 8 N-tiles.
// ---------------------------------------------------------------------------
__global__ __launch_bounds__(256) void outproj_kernel(const _Float16* __restrict__ att_out,
                                                      const float* __restrict__ W_out,
                                                      const float* __restrict__ b_out,
                                                      float* __restrict__ out) {
  const int lane = threadIdx.x & 31;
  const int wid  = threadIdx.x >> 5;
  const int wg   = blockIdx.x * 8 + wid;   // 0..32767
  const int rb   = wg >> 2;                // row block 0..8191
  const int cg   = wg & 3;                 // col group (128 cols each)
  const int row0 = rb * 16;
  const int col0 = cg * 128;
  const int m    = lane & 15;
  const int hg   = lane >> 4;

  // A: 16 rows x 256 (8 K-chunks of 32), f16 already
  v16h a[8];
  const _Float16* ap = att_out + (size_t)(row0 + m) * CDIM;
  for (int kc = 0; kc < 8; ++kc)
    for (int e = 0; e < 16; ++e)
      a[kc][e] = ap[kc * 32 + a_k(e, hg)];

  for (int nt = 0; nt < 8; ++nt) {
    const int colN = col0 + nt * 16 + m;
    v8f dacc = {};
    for (int kc = 0; kc < 8; ++kc) {
      v16h bw;
      for (int e = 0; e < 16; ++e) {
        int k = kc * 32 + b_k(e, hg);
        bw[e] = (_Float16)W_out[(size_t)k * TWO_C + colN];
      }
      dacc = WMMA_F32_F16(a[kc], bw, dacc);
    }
    const float bias = b_out[colN];
    for (int r = 0; r < 8; ++r) {
      int row = row0 + r + 8 * hg;
      out[(size_t)row * TWO_C + colN] = dacc[r] + bias;
    }
  }
}

// ---------------------------------------------------------------------------
extern "C" void kernel_launch(void* const* d_in, const int* in_sizes, int n_in,
                              void* d_out, int out_size, void* d_ws, size_t ws_size,
                              hipStream_t stream) {
  (void)in_sizes; (void)n_in; (void)out_size; (void)ws_size;

  const float*         x      = (const float*)d_in[0];
  const unsigned char* mask   = (const unsigned char*)d_in[1];
  const float*         emb_q  = (const float*)d_in[2];
  const float*         W_emb  = (const float*)d_in[3];
  const float*         b_emb  = (const float*)d_in[4];
  const float*         W_gate = (const float*)d_in[5];
  const float*         b_gate = (const float*)d_in[6];
  const float*         W_out  = (const float*)d_in[7];
  const float*         b_out  = (const float*)d_in[8];

  float* out        = (float*)d_out;
  float* mask_out_f = out + (size_t)NROWS * TWO_C;   // second tuple output

  // workspace layout
  char* ws = (char*)d_ws;
  _Float16*      Qh      = (_Float16*)ws;                               // 64 MiB
  _Float16*      att_o   = (_Float16*)(ws + (size_t)64 * 1024 * 1024);  // 64 MiB
  unsigned char* maskm   = (unsigned char*)(ws + (size_t)128 * 1024 * 1024);

  mask_kernel   <<<NGRP,         64, 0, stream>>>(mask, maskm, mask_out_f);
  qproj_kernel  <<<NROWS/16/8,  256, 0, stream>>>(emb_q, W_emb, b_emb, W_gate, b_gate, Qh);
  attn_kernel   <<<NGRP*NHEAD,   32, 0, stream>>>(x, Qh, maskm, att_o);
  outproj_kernel<<<4096,        256, 0, stream>>>(att_o, W_out, b_out, out);
}
